// MambaAudioModel_19945828122793
// MI455X (gfx1250) — matrix operand order repaired and checked by reading, not compiled, not run on hardware
//
#include <hip/hip_runtime.h>
#include <hip/hip_bf16.h>

// ---------------------------------------------------------------- constants
constexpr int kE  = 512;    // n_embed
constexpr int kDI = 1024;   // d_inner
constexpr int kNS = 16;     // d_state
constexpr int kDC = 4;      // d_conv
constexpr int kR  = 32;     // dt_rank
constexpr int kL  = 4;      // layers
constexpr int kV  = 1024;   // vocab
constexpr int kT  = 2048;   // seq len
constexpr int kB  = 4;      // batch
constexpr int kBT = kB * kT;           // 8192 GEMM rows
constexpr int kXP = kR + 2 * kNS;      // 64

// ---------------------------------------------------------------- types
typedef __attribute__((ext_vector_type(16))) __bf16        v16bf;
typedef __attribute__((ext_vector_type(8)))  unsigned int  v8u;
typedef __attribute__((ext_vector_type(8)))  float         v8f;
typedef int v4i_gcc __attribute__((vector_size(16)));      // matches builtin's V4i param

#define GAS __attribute__((address_space(1)))
#define LAS __attribute__((address_space(3)))

__device__ __forceinline__ unsigned short f2bf(float f) {
    unsigned int u = __builtin_bit_cast(unsigned int, f);
    u += 0x7FFFu + ((u >> 16) & 1u);      // round-to-nearest-even
    return (unsigned short)(u >> 16);
}
__device__ __forceinline__ float bf2f(unsigned short h) {
    return __builtin_bit_cast(float, ((unsigned int)h) << 16);
}
__device__ __forceinline__ float silu_f(float x) { return x / (1.f + __expf(-x)); }

// ---------------------------------------------------------------- async global->LDS copy (CDNA5 async path, ASYNCcnt)
#if __has_builtin(__builtin_amdgcn_global_load_async_to_lds_b128)
#define HAVE_ASYNC_LDS 1
#else
#define HAVE_ASYNC_LDS 0
#endif

__device__ __forceinline__ void copy16_to_lds(const unsigned short* g, unsigned short* l) {
#if HAVE_ASYNC_LDS
    __builtin_amdgcn_global_load_async_to_lds_b128(
        (GAS v4i_gcc*)g, (LAS v4i_gcc*)l, 0, 0);
#else
    *(uint4*)l = *(const uint4*)g;
#endif
}
__device__ __forceinline__ void wait_tiles() {
#if HAVE_ASYNC_LDS
#if __has_builtin(__builtin_amdgcn_s_wait_asynccnt)
    __builtin_amdgcn_s_wait_asynccnt(0);
#else
    asm volatile("s_wait_asynccnt 0x0" ::: "memory");
#endif
#endif
}

// ---------------------------------------------------------------- f32 -> bf16 cast
__global__ __launch_bounds__(256) void cvt_bf16_kernel(const float* __restrict__ in,
                                                       unsigned short* __restrict__ out, int n) {
    int i = blockIdx.x * 256 + threadIdx.x;
    if (i < n) out[i] = f2bf(in[i]);
}

// ---------------------------------------------------------------- embedding (f32 residual stream)
__global__ __launch_bounds__(256) void embed_kernel(const int* __restrict__ idx,
                                                    const float* __restrict__ tok,
                                                    const float* __restrict__ pos,
                                                    float* __restrict__ x) {
    int gid = blockIdx.x * 256 + threadIdx.x;          // over BT*E
    if (gid >= kBT * kE) return;
    int e  = gid % kE;
    int bt = gid / kE;
    int t  = bt % kT;
    x[gid] = tok[(size_t)idx[bt] * kE + e] + pos[(size_t)t * kE + e];
}

// ---------------------------------------------------------------- layernorm (block per row) -> bf16
__global__ __launch_bounds__(256) void layernorm_kernel(const float* __restrict__ x,
                                                        const float* __restrict__ g,
                                                        const float* __restrict__ b,
                                                        unsigned short* __restrict__ o) {
    __shared__ float red[256];
    const int tid = threadIdx.x;
    const float* xr = x + (size_t)blockIdx.x * kE;
    float s = 0.f;
    for (int e = tid; e < kE; e += 256) s += xr[e];
    red[tid] = s; __syncthreads();
    for (int off = 128; off > 0; off >>= 1) {
        if (tid < off) red[tid] += red[tid + off];
        __syncthreads();
    }
    const float mean = red[0] * (1.f / kE);
    __syncthreads();
    float v = 0.f;
    for (int e = tid; e < kE; e += 256) { float d = xr[e] - mean; v += d * d; }
    red[tid] = v; __syncthreads();
    for (int off = 128; off > 0; off >>= 1) {
        if (tid < off) red[tid] += red[tid + off];
        __syncthreads();
    }
    const float rstd = rsqrtf(red[0] * (1.f / kE) + 1e-5f);
    unsigned short* orow = o + (size_t)blockIdx.x * kE;
    for (int e = tid; e < kE; e += 256)
        orow[e] = f2bf((xr[e] - mean) * rstd * g[e] + b[e]);
}

// ---------------------------------------------------------------- WMMA GEMM with async double-buffered LDS pipeline
// A: MxK bf16, row stride lda.  W: NxK bf16 row-major.  C: MxN (f32 or bf16).
// act: 0=none 1=relu 2=softplus
#define BM 128
#define BN 64
#define BK 32
constexpr int BKP = BK + 8;     // bf16 row stride in LDS (80B: 16B-aligned, bank-spread)
constexpr int RSU = BKP / 2;    // row stride in uints

__global__ __launch_bounds__(256)
void gemm_bf16_wmma(const unsigned short* __restrict__ A, int lda,
                    const unsigned short* __restrict__ W,
                    void* __restrict__ Cv,
                    const float* __restrict__ bias,
                    const float* __restrict__ residual,
                    int M, int N, int K, int act, int c_bf16) {
    __shared__ unsigned short As[2][BM][BKP];
    __shared__ unsigned short Bs[2][BN][BKP];

    const int tid  = threadIdx.x;
    const int bm   = blockIdx.y * BM;
    const int bn   = blockIdx.x * BN;
    const int wave = tid >> 5;
    const int lane = tid & 31;
    const int wm   = (wave >> 1) * 32;   // 4 waves along M
    const int wn   = (wave & 1) * 32;    // 2 waves along N
    const int lr   = lane & 15;
    const int hi   = lane >> 4;

    // stage one (A,B) K-tile pair into LDS buffer `buf` via async b128 copies
    auto prefetch = [&](int buf, int kt) {
        #pragma unroll
        for (int i = 0; i < 2; ++i) {                 // A: 128x32 bf16 = 512 x 16B chunks
            int id = tid + 256 * i;
            int r = id >> 2, ch = id & 3;
            copy16_to_lds(A + (size_t)(bm + r) * lda + kt + ch * 8, &As[buf][r][ch * 8]);
        }
        {                                             // B: 64x32 bf16 = 256 x 16B chunks
            int r = tid >> 2, ch = tid & 3;
            copy16_to_lds(W + (size_t)(bn + r) * K + kt + ch * 8, &Bs[buf][r][ch * 8]);
        }
    };

    v8f acc[2][2] = {};
    int buf = 0;
    prefetch(0, 0);

    for (int kt = 0; kt < K; kt += BK) {
        wait_tiles();                 // own async copies into `buf` done
        __syncthreads();              // everyone's copies visible
        if (kt + BK < K) prefetch(buf ^ 1, kt + BK);   // overlap next tile with compute

        const unsigned int* Ap = (const unsigned int*)&As[buf][0][0];
        const unsigned int* Bp = (const unsigned int*)&Bs[buf][0][0];

        // fragments: packed uint pairs match the CDNA5 bf16 A(16x32)/B(32x16) lane layouts
        v8u af[2], bfr[2];
        #pragma unroll
        for (int mi = 0; mi < 2; ++mi) {
            const int ro = (wm + mi * 16 + lr) * RSU;
            #pragma unroll
            for (int p = 0; p < 4; ++p) {
                af[mi][p]     = Ap[ro + 4 * hi + p];
                af[mi][4 + p] = Ap[ro + 8 + 4 * hi + p];
            }
        }
        #pragma unroll
        for (int ni = 0; ni < 2; ++ni) {
            const int co = (wn + ni * 16 + lr) * RSU;
            #pragma unroll
            for (int p = 0; p < 4; ++p) {
                bfr[ni][p]     = Bp[co + 4 * hi + p];
                bfr[ni][4 + p] = Bp[co + 8 + 4 * hi + p];
            }
        }
        #pragma unroll
        for (int mi = 0; mi < 2; ++mi)
            #pragma unroll
            for (int ni = 0; ni < 2; ++ni)
                acc[mi][ni] = __builtin_amdgcn_wmma_f32_16x16x32_bf16(
                    false, __builtin_bit_cast(v16bf, af[mi]),
                    false, __builtin_bit_cast(v16bf, bfr[ni]),
                    (short)0, acc[mi][ni], false, false);

        __syncthreads();              // all waves done reading `buf` before it is refilled
        buf ^= 1;
    }

    // epilogue: bias -> activation -> residual -> store (f32 C/D lane layout)
    float* Cf = (float*)Cv;
    unsigned short* Cb = (unsigned short*)Cv;
    #pragma unroll
    for (int mi = 0; mi < 2; ++mi) {
        #pragma unroll
        for (int ni = 0; ni < 2; ++ni) {
            const int nG = bn + wn + ni * 16 + lr;
            const float bv = bias ? bias[nG] : 0.f;
            #pragma unroll
            for (int v = 0; v < 8; ++v) {
                const int mG = bm + wm + mi * 16 + v + 8 * hi;
                float val = acc[mi][ni][v] + bv;
                if (act == 1) {
                    val = val > 0.f ? val : 0.f;
                } else if (act == 2) {
                    float sp = __logf(1.f + __expf(val));
                    val = (val > 20.f) ? val : sp;
                }
                const size_t o = (size_t)mG * N + nG;
                if (residual) val += residual[o];
                if (c_bf16) Cb[o] = f2bf(val); else Cf[o] = val;
            }
        }
    }
}

// ---------------------------------------------------------------- causal depthwise conv + SiLU (bf16 in/out)
__global__ __launch_bounds__(256) void conv_silu_kernel(const unsigned short* __restrict__ xz,
                                                        const float* __restrict__ cw,
                                                        const float* __restrict__ cb,
                                                        unsigned short* __restrict__ xs) {
    int gid = blockIdx.x * 256 + threadIdx.x;          // over BT*DI
    if (gid >= kBT * kDI) return;
    const int d  = gid % kDI;
    const int bt = gid / kDI;
    const int t  = bt % kT;
    const int b  = bt / kT;
    float acc = cb[d];
    #pragma unroll
    for (int k = 0; k < kDC; ++k) {
        const int tt = t - (kDC - 1) + k;
        if (tt >= 0)
            acc += cw[d * kDC + k] * bf2f(xz[((size_t)b * kT + tt) * (2 * kDI) + d]);
    }
    xs[gid] = f2bf(silu_f(acc));
}

// ---------------------------------------------------------------- selective scan (thread per (b,d), states in regs)
__global__ __launch_bounds__(256) void scan_kernel(const unsigned short* __restrict__ u,
                                                   const float* __restrict__ dt,
                                                   const float* __restrict__ A_log,
                                                   const unsigned short* __restrict__ xdbl,
                                                   const float* __restrict__ Dp,
                                                   float* __restrict__ y) {
    const int gid = blockIdx.x * 256 + threadIdx.x;    // B*DI
    if (gid >= kB * kDI) return;
    const int b = gid / kDI;
    const int d = gid % kDI;

    float Arow[kNS], h[kNS];
    #pragma unroll
    for (int n = 0; n < kNS; ++n) {
        Arow[n] = -__expf(A_log[d * kNS + n]);
        h[n] = 0.f;
    }
    const float Dd = Dp[d];

    for (int t = 0; t < kT; ++t) {
        const size_t bt = (size_t)b * kT + t;
        if (t + 8 < kT) {                              // hide latency of the serial chain
            __builtin_prefetch(&u[(bt + 8) * kDI + d], 0, 1);
            __builtin_prefetch(&dt[(bt + 8) * kDI + d], 0, 1);
        }
        const float dtv = dt[bt * kDI + d];
        const float uv  = bf2f(u[bt * kDI + d]);
        const float du  = dtv * uv;
        const unsigned short* xr = xdbl + bt * kXP;
        float acc = 0.f;
        #pragma unroll
        for (int n = 0; n < kNS; ++n) {
            const float dA = __expf(dtv * Arow[n]);
            h[n] = dA * h[n] + du * bf2f(xr[kR + n]);          // B_t
            acc += h[n] * bf2f(xr[kR + kNS + n]);              // C_t
        }
        y[bt * kDI + d] = acc + uv * Dd;
    }
}

// ---------------------------------------------------------------- gate: g = bf16(y * silu(z))
__global__ __launch_bounds__(256) void gate_kernel(const float* __restrict__ y,
                                                   const unsigned short* __restrict__ xz,
                                                   unsigned short* __restrict__ g) {
    int gid = blockIdx.x * 256 + threadIdx.x;          // over BT*DI
    if (gid >= kBT * kDI) return;
    const int d  = gid % kDI;
    const int bt = gid / kDI;
    g[gid] = f2bf(y[gid] * silu_f(bf2f(xz[(size_t)bt * (2 * kDI) + kDI + d])));
}

// ================================================================ host
extern "C" void kernel_launch(void* const* d_in, const int* in_sizes, int n_in,
                              void* d_out, int out_size, void* d_ws, size_t ws_size,
                              hipStream_t stream) {
    (void)in_sizes; (void)n_in; (void)out_size; (void)ws_size;

    const int*   idx     = (const int*)  d_in[0];
    const float* tok_emb = (const float*)d_in[1];
    const float* pos_emb = (const float*)d_in[2];
    const float* ln1_g   = (const float*)d_in[3];
    const float* ln1_b   = (const float*)d_in[4];
    const float* ln2_g   = (const float*)d_in[5];
    const float* ln2_b   = (const float*)d_in[6];
    const float* W_in    = (const float*)d_in[7];
    const float* conv_w  = (const float*)d_in[8];
    const float* conv_b  = (const float*)d_in[9];
    const float* W_xp    = (const float*)d_in[10];
    const float* W_dt    = (const float*)d_in[11];
    const float* b_dt    = (const float*)d_in[12];
    const float* A_log   = (const float*)d_in[13];
    const float* D_ssm   = (const float*)d_in[14];
    const float* W_out   = (const float*)d_in[15];
    const float* W1      = (const float*)d_in[16];
    const float* b1      = (const float*)d_in[17];
    const float* W2      = (const float*)d_in[18];
    const float* b2      = (const float*)d_in[19];
    const float* W_lm    = (const float*)d_in[20];
    const float* b_lm    = (const float*)d_in[21];

    // -------- workspace bump allocator
    char* ws = (char*)d_ws;
    size_t off = 0;
    auto alloc = [&](size_t bytes) -> void* {
        size_t a = (off + 255) & ~(size_t)255;
        off = a + bytes;
        return (void*)(ws + a);
    };
    float*          x     = (float*)         alloc((size_t)kBT * kE * 4);
    unsigned short* xnb   = (unsigned short*)alloc((size_t)kBT * kE * 2);
    unsigned short* xzb   = (unsigned short*)alloc((size_t)kBT * 2 * kDI * 2); // alias: MLP hidden
    unsigned short* xsb   = (unsigned short*)alloc((size_t)kBT * kDI * 2);
    unsigned short* xdblb = (unsigned short*)alloc((size_t)kBT * kXP * 2);
    float*          dtb   = (float*)         alloc((size_t)kBT * kDI * 4);
    float*          yb    = (float*)         alloc((size_t)kBT * kDI * 4);
    unsigned short* gbf   = (unsigned short*)alloc((size_t)kBT * kDI * 2);
    unsigned short* xbf   = (unsigned short*)alloc((size_t)kBT * kE * 2);
    unsigned short* hmlpb = xzb;                      // xz dead after gate; 4E == 2*DI

    const size_t nWin = (size_t)kL * 2 * kDI * kE;
    const size_t nWxp = (size_t)kL * kXP * kDI;
    const size_t nWdt = (size_t)kL * kDI * kR;
    const size_t nWo  = (size_t)kL * kE * kDI;
    const size_t nW1  = (size_t)kL * 4 * kE * kE;
    const size_t nW2  = (size_t)kL * kE * 4 * kE;
    const size_t nWlm = (size_t)kV * kE;
    unsigned short* Winb = (unsigned short*)alloc(nWin * 2);
    unsigned short* Wxpb = (unsigned short*)alloc(nWxp * 2);
    unsigned short* Wdtb = (unsigned short*)alloc(nWdt * 2);
    unsigned short* Wob  = (unsigned short*)alloc(nWo  * 2);
    unsigned short* W1b  = (unsigned short*)alloc(nW1  * 2);
    unsigned short* W2b  = (unsigned short*)alloc(nW2  * 2);
    unsigned short* Wlmb = (unsigned short*)alloc(nWlm * 2);

    auto cvt = [&](const float* src, unsigned short* dst, size_t n) {
        cvt_bf16_kernel<<<dim3((unsigned)((n + 255) / 256)), dim3(256), 0, stream>>>(src, dst, (int)n);
    };
    cvt(W_in, Winb, nWin); cvt(W_xp, Wxpb, nWxp); cvt(W_dt, Wdtb, nWdt);
    cvt(W_out, Wob, nWo);  cvt(W1, W1b, nW1);     cvt(W2, W2b, nW2);
    cvt(W_lm, Wlmb, nWlm);

    auto gemm = [&](const unsigned short* A, int lda, const unsigned short* W, void* C,
                    const float* bias, const float* res, int M, int N, int K, int act, int cbf) {
        gemm_bf16_wmma<<<dim3(N / BN, M / BM), dim3(256), 0, stream>>>(
            A, lda, W, C, bias, res, M, N, K, act, cbf);
    };

    // -------- embedding
    embed_kernel<<<dim3((kBT * kE + 255) / 256), dim3(256), 0, stream>>>(idx, tok_emb, pos_emb, x);

    for (int l = 0; l < kL; ++l) {
        // --- Mamba block
        layernorm_kernel<<<dim3(kBT), dim3(256), 0, stream>>>(x, ln1_g + l * kE, ln1_b + l * kE, xnb);
        gemm(xnb, kE, Winb + (size_t)l * 2 * kDI * kE, xzb, nullptr, nullptr,
             kBT, 2 * kDI, kE, 0, 1 /*bf16 out*/);
        conv_silu_kernel<<<dim3((kBT * kDI + 255) / 256), dim3(256), 0, stream>>>(
            xzb, conv_w + (size_t)l * kDI * kDC, conv_b + (size_t)l * kDI, xsb);
        gemm(xsb, kDI, Wxpb + (size_t)l * kXP * kDI, xdblb, nullptr, nullptr,
             kBT, kXP, kDI, 0, 1 /*bf16 out*/);
        gemm(xdblb, kXP, Wdtb + (size_t)l * kDI * kR, dtb, b_dt + (size_t)l * kDI, nullptr,
             kBT, kDI, kR, 2 /*softplus*/, 0);
        scan_kernel<<<dim3((kB * kDI + 255) / 256), dim3(256), 0, stream>>>(
            xsb, dtb, A_log + (size_t)l * kDI * kNS, xdblb, D_ssm + (size_t)l * kDI, yb);
        gate_kernel<<<dim3((kBT * kDI + 255) / 256), dim3(256), 0, stream>>>(yb, xzb, gbf);
        gemm(gbf, kDI, Wob + (size_t)l * kE * kDI, x, nullptr, x /*residual*/,
             kBT, kE, kDI, 0, 0);

        // --- MLP block
        layernorm_kernel<<<dim3(kBT), dim3(256), 0, stream>>>(x, ln2_g + l * kE, ln2_b + l * kE, xnb);
        gemm(xnb, kE, W1b + (size_t)l * 4 * kE * kE, hmlpb, b1 + (size_t)l * 4 * kE, nullptr,
             kBT, 4 * kE, kE, 1 /*relu*/, 1 /*bf16 out*/);
        gemm(hmlpb, 4 * kE, W2b + (size_t)l * kE * 4 * kE, x, b2 + (size_t)l * kE, x /*residual*/,
             kBT, kE, 4 * kE, 0, 0);
    }

    // -------- LM head
    cvt_bf16_kernel<<<dim3((kBT * kE + 255) / 256), dim3(256), 0, stream>>>(x, xbf, kBT * kE);
    gemm(xbf, kE, Wlmb, (float*)d_out, b_lm, nullptr, kBT, kV, kE, 0, 0);
}
